// EdgeClassifier_10393820857055
// MI455X (gfx1250) — matrix operand order repaired and checked by reading, not compiled, not run on hardware
//
#include <hip/hip_runtime.h>
#include <stdint.h>

typedef __attribute__((ext_vector_type(16))) __bf16 bf16x16;
typedef __attribute__((ext_vector_type(8)))  float  f32x8;

union FragBF {
  bf16x16 bf;
  uint32_t u[8];
};

#define NWAVES 8
#define THREADS (NWAVES * 32)
#define MT 2                 // M-tiles per wave (32 edges per wave-tile)
#define RS1  264             // padded LDS row stride in bf16 elems (256 + 8)
#define RS1U (RS1 / 2)       // 132 dwords per row

// LDS byte offsets (all 16B aligned)
#define LDS_W1      0
#define LDS_W1_SZ   (256 * RS1 * 2)          // 135168
#define LDS_W2      (LDS_W1 + LDS_W1_SZ)
#define LDS_W2_SZ   (32 * RS1 * 2)           // 16896
#define LDS_B1      (LDS_W2 + LDS_W2_SZ)     // 152064
#define LDS_B2      (LDS_B1 + 256 * 4)       // 153088
#define LDS_STAGE   (LDS_B2 + 32 * 4)        // 153216
#define STAGE_PER_WAVE (MT * 16 * RS1 * 2)   // shared A/H stage = 16896
#define LDS_TOTAL   (LDS_STAGE + NWAVES * STAGE_PER_WAVE)  // 288384 <= 320KB

__device__ __forceinline__ uint32_t pk_bf16(float a, float b) {
  uint32_t xa = __float_as_uint(a);
  uint32_t xb = __float_as_uint(b);
  xa = (xa + 0x7FFFu + ((xa >> 16) & 1u)) >> 16;   // round-to-nearest-even
  xb = (xb + 0x7FFFu + ((xb >> 16) & 1u)) >> 16;
  return (xa & 0xFFFFu) | (xb << 16);
}

__device__ __forceinline__ uint16_t cvt_bf16(float a) {
  uint32_t x = __float_as_uint(a);
  return (uint16_t)((x + 0x7FFFu + ((x >> 16) & 1u)) >> 16);
}

__global__ void __launch_bounds__(THREADS)
edge_mlp_wmma(const float* __restrict__ emb,
              const long long* __restrict__ eidx,   // [2, E]
              const float* __restrict__ W1, const float* __restrict__ b1,
              const float* __restrict__ W2, const float* __restrict__ b2,
              float* __restrict__ out,
              int nEdges, int nTiles, int waveStride)
{
  extern __shared__ char smem[];
  uint32_t* w1u = (uint32_t*)(smem + LDS_W1);   // bf16 [256][RS1], dword view
  uint32_t* w2u = (uint32_t*)(smem + LDS_W2);   // bf16 [32][RS1]
  float*    b1l = (float*)(smem + LDS_B1);
  float*    b2l = (float*)(smem + LDS_B2);

  const int tid  = threadIdx.x;
  const int lane = tid & 31;
  const int wave = tid >> 5;

  // One staging buffer per wave, reused: first as gathered-A, then as hidden-H.
  uint16_t* St = (uint16_t*)(smem + LDS_STAGE + wave * STAGE_PER_WAVE);

  // ---------- Phase 0: cooperative f32 -> bf16 weight/bias staging ----------
  {
    const float4* w1v = (const float4*)W1;                 // [256][256] -> 16384 vec4
    for (int i = tid; i < (256 * 256) / 4; i += THREADS) {
      float4 v = w1v[i];
      int row = i >> 6, c = (i & 63) * 2;
      uint32_t* d = w1u + row * RS1U + c;
      d[0] = pk_bf16(v.x, v.y);
      d[1] = pk_bf16(v.z, v.w);
    }
    const float4* w2v = (const float4*)W2;                 // [32][256] -> 2048 vec4
    for (int i = tid; i < (32 * 256) / 4; i += THREADS) {
      float4 v = w2v[i];
      int row = i >> 6, c = (i & 63) * 2;
      uint32_t* d = w2u + row * RS1U + c;
      d[0] = pk_bf16(v.x, v.y);
      d[1] = pk_bf16(v.z, v.w);
    }
    for (int i = tid; i < 256; i += THREADS) b1l[i] = b1[i];
    for (int i = tid; i < 32;  i += THREADS) b2l[i] = b2[i];
  }
  __syncthreads();

  const int m     = lane & 15;   // A-fragment row / B-fragment column
  const int khalf = lane >> 4;   // lane half selects K sub-range
  const float4* embv = (const float4*)emb;

  // ---------- Phase 1: per-wave 32-edge tiles (2 M-subtiles), grid-stride ----
  for (int tile = blockIdx.x * NWAVES + wave; tile < nTiles; tile += waveStride) {
    const int ebase = tile * (MT * 16);

    // Gather: 32 edges x {src,dst} = 64 node rows of 128 f32; one b128/lane/row.
    #pragma unroll 4
    for (int r = 0; r < 64; ++r) {
      int e    = ebase + (r >> 1);
      int half = r & 1;
      int ec   = e < nEdges ? e : (nEdges - 1);
      long long node = eidx[(size_t)half * nEdges + ec];
      float4 v = embv[(size_t)node * 32 + lane];
      uint32_t* d = (uint32_t*)(St + (r >> 1) * RS1 + half * 128 + lane * 4);
      d[0] = pk_bf16(v.x, v.y);
      d[1] = pk_bf16(v.z, v.w);
    }

    // Load both 16x256 bf16 A-subtiles into registers (WMMA A-fragment layout).
    // Lane L (0-15): row M=L; VGPR j in [0,3]: K = 32*kb + khalf*8 + 2j,+1 ;
    // VGPR j in [4,7]: K += 16.  (contiguous 16B chunks -> ds b128 loads)
    FragBF a[MT][8];
    #pragma unroll
    for (int mt = 0; mt < MT; ++mt) {
      const uint32_t* Arow = (const uint32_t*)(St + (mt * 16 + m) * RS1);
      #pragma unroll
      for (int kb = 0; kb < 8; ++kb) {
        int k0 = kb * 32 + khalf * 8;
        #pragma unroll
        for (int j = 0; j < 4; ++j) {
          a[mt][kb].u[j]     = Arow[(k0 >> 1) + j];
          a[mt][kb].u[4 + j] = Arow[((k0 + 16) >> 1) + j];
        }
      }
    }

    // GEMM1: hidden = relu(A @ W1^T + b1); each B-fragment feeds both M-subtiles.
    // B-fragments are double-buffered so the dscnt wait covers only the older
    // 4-load group (latency overlapped with the WMMA pair).
    // A-fragments are register-resident, so St is dead -> reuse it as H stage.
    #pragma unroll 2
    for (int nt = 0; nt < 16; ++nt) {
      int n = nt * 16 + m;
      // B layout: lanes 0-15 hold K=0..7,16..23, lanes 16-31 hold K=8..15,24..31
      const uint32_t* Brow = w1u + n * RS1U + khalf * 4;
      FragBF bcur, bnxt;
      #pragma unroll
      for (int j = 0; j < 8; ++j) bcur.u[j] = Brow[j + (j >> 2) * 4];
      f32x8 acc0 = {0.f, 0.f, 0.f, 0.f, 0.f, 0.f, 0.f, 0.f};
      f32x8 acc1 = {0.f, 0.f, 0.f, 0.f, 0.f, 0.f, 0.f, 0.f};
      #pragma unroll
      for (int kb = 0; kb < 8; ++kb) {
        if (kb < 7) {
          #pragma unroll
          for (int j = 0; j < 8; ++j) bnxt.u[j] = Brow[(kb + 1) * 16 + j + (j >> 2) * 4];
        }
        acc0 = __builtin_amdgcn_wmma_f32_16x16x32_bf16(
                   false, a[0][kb].bf, false, bcur.bf, (short)0, acc0, false, false);
        acc1 = __builtin_amdgcn_wmma_f32_16x16x32_bf16(
                   false, a[1][kb].bf, false, bcur.bf, (short)0, acc1, false, false);
        bcur = bnxt;
      }
      float bias = b1l[n];
      #pragma unroll
      for (int r = 0; r < 8; ++r) {   // D layout: vgpr r -> M = r + 8*khalf, N = lane&15
        float v0 = acc0[r] + bias;
        float v1 = acc1[r] + bias;
        v0 = v0 > 0.f ? v0 : 0.f;
        v1 = v1 > 0.f ? v1 : 0.f;
        St[(r + khalf * 8) * RS1 + n]        = cvt_bf16(v0);
        St[(16 + r + khalf * 8) * RS1 + n]   = cvt_bf16(v1);
      }
    }

    // Re-load hidden tiles in A-fragment layout (cross-lane relayout via LDS).
    FragBF h[MT][8];
    #pragma unroll
    for (int mt = 0; mt < MT; ++mt) {
      const uint32_t* Hrow = (const uint32_t*)(St + (mt * 16 + m) * RS1);
      #pragma unroll
      for (int kb = 0; kb < 8; ++kb) {
        int k0 = kb * 32 + khalf * 8;
        #pragma unroll
        for (int j = 0; j < 4; ++j) {
          h[mt][kb].u[j]     = Hrow[(k0 >> 1) + j];
          h[mt][kb].u[4 + j] = Hrow[((k0 + 16) >> 1) + j];
        }
      }
    }

    // GEMM2: out = hidden @ W2^T + b2; fused f32 store.
    #pragma unroll
    for (int nt = 0; nt < 2; ++nt) {
      int n = nt * 16 + m;
      const uint32_t* Brow = w2u + n * RS1U + khalf * 4;
      FragBF bcur, bnxt;
      #pragma unroll
      for (int j = 0; j < 8; ++j) bcur.u[j] = Brow[j + (j >> 2) * 4];
      f32x8 acc0 = {0.f, 0.f, 0.f, 0.f, 0.f, 0.f, 0.f, 0.f};
      f32x8 acc1 = {0.f, 0.f, 0.f, 0.f, 0.f, 0.f, 0.f, 0.f};
      #pragma unroll
      for (int kb = 0; kb < 8; ++kb) {
        if (kb < 7) {
          #pragma unroll
          for (int j = 0; j < 8; ++j) bnxt.u[j] = Brow[(kb + 1) * 16 + j + (j >> 2) * 4];
        }
        acc0 = __builtin_amdgcn_wmma_f32_16x16x32_bf16(
                   false, h[0][kb].bf, false, bcur.bf, (short)0, acc0, false, false);
        acc1 = __builtin_amdgcn_wmma_f32_16x16x32_bf16(
                   false, h[1][kb].bf, false, bcur.bf, (short)0, acc1, false, false);
        bcur = bnxt;
      }
      float bias = b2l[n];
      #pragma unroll
      for (int r = 0; r < 8; ++r) {
        int e0 = ebase + r + khalf * 8;
        int e1 = e0 + 16;
        if (e0 < nEdges) out[(size_t)e0 * 32 + n] = acc0[r] + bias;
        if (e1 < nEdges) out[(size_t)e1 * 32 + n] = acc1[r] + bias;
      }
    }
  }
}

extern "C" void kernel_launch(void* const* d_in, const int* in_sizes, int n_in,
                              void* d_out, int out_size, void* d_ws, size_t ws_size,
                              hipStream_t stream) {
  (void)n_in; (void)out_size; (void)d_ws; (void)ws_size;
  const float*     emb  = (const float*)d_in[0];
  const long long* eidx = (const long long*)d_in[1];   // int64 [2, E]
  const float*     W1   = (const float*)d_in[2];
  const float*     b1   = (const float*)d_in[3];
  const float*     W2   = (const float*)d_in[4];
  const float*     b2   = (const float*)d_in[5];
  float*           out  = (float*)d_out;

  int nEdges = in_sizes[1] / 2;
  int nTiles = (nEdges + (MT * 16) - 1) / (MT * 16);

  int blocks = 512;
  int maxBlocks = (nTiles + NWAVES - 1) / NWAVES;
  if (blocks > maxBlocks) blocks = maxBlocks;
  int waveStride = blocks * NWAVES;

  hipFuncSetAttribute((const void*)edge_mlp_wmma,
                      hipFuncAttributeMaxDynamicSharedMemorySize, LDS_TOTAL);
  edge_mlp_wmma<<<blocks, THREADS, LDS_TOTAL, stream>>>(
      emb, eidx, W1, b1, W2, b2, out, nEdges, nTiles, waveStride);
}
